// Seq2SeqGenerator_41901700940208
// MI455X (gfx1250) — matrix operand order repaired and checked by reading, not compiled, not run on hardware
//
#include <hip/hip_runtime.h>

// Seq2SeqGenerator LSTM: B=512, LAT=64, COND=32, H=256, L=2, OUT=2, T=200
#define BQ    512
#define LATQ  64
#define CONDQ 32
#define CTXQ  96
#define HQ    256
#define LQ    2
#define OUTQ  2
#define GW    1024      // 4*H gate width
#define BT    16        // batch rows per workgroup
#define NT    512       // threads per block = 16 waves
#define XK    128       // padded K for input projection (OUT+CTX=98 -> 128)
#define INST  136       // inbuf row stride (bf16 elems), 16B aligned, bank-friendly
#define HST   264       // x/h buffer row stride (bf16 elems), 16B aligned, bank-friendly

typedef __attribute__((ext_vector_type(16))) __bf16 v16bf;
typedef __attribute__((ext_vector_type(8)))  __bf16 v8bf;
typedef __attribute__((ext_vector_type(8)))  float  v8f;

__device__ __forceinline__ float fexp2(float x) { return __builtin_amdgcn_exp2f(x); }
__device__ __forceinline__ float sigm(float x)  { return 1.0f / (1.0f + fexp2(-1.44269504f * x)); }
__device__ __forceinline__ float tanh_(float x) { return 1.0f - 2.0f / (1.0f + fexp2(2.88539008f * x)); }

__device__ __forceinline__ v8f wmma_bf16(v16bf a, v16bf b, v8f c) {
    // D = A(16x32 bf16) x B(32x16 bf16) + C(16x16 f32)
    return __builtin_amdgcn_wmma_f32_16x16x32_bf16(false, a, false, b, (short)0, c, false, false);
}

// Opaque zero offset: defeats cross-timestep LICM of the (invariant) weight loads
// without destroying the global address-space provenance of the kernarg pointer,
// so weight fetches stay global_load_b128 (saddr+voffset) re-served from L2.
__device__ __forceinline__ int opaque_zero() {
    int x = 0;
    asm volatile("" : "+v"(x));
    return x;
}

// A-fragment (16x32 bf16) from a precomputed per-lane base:
// elements 0..15 = K[k0+c0 .. +15] (c0 = half*16).
__device__ __forceinline__ v16bf loadA(const __bf16* p) {
    v8bf lo = *(const v8bf*)(p);
    v8bf hi = *(const v8bf*)(p + 8);
    return __builtin_shufflevector(lo, hi, 0,1,2,3,4,5,6,7,8,9,10,11,12,13,14,15);
}

// ---------------- prologue: fp32 -> bf16 weight conversion ----------------
__global__ void convert_weights(const float* __restrict__ Wih, const float* __restrict__ Whh,
                                const float* __restrict__ Win,
                                __bf16* __restrict__ Wih_b, __bf16* __restrict__ Whh_b,
                                __bf16* __restrict__ Win_b) {
    int i = blockIdx.x * blockDim.x + threadIdx.x;
    const int NW = LQ * GW * HQ;  // 524288
    if (i < NW) {
        Wih_b[i] = (__bf16)Wih[i];
    } else if (i < 2 * NW) {
        Whh_b[i - NW] = (__bf16)Whh[i - NW];
    } else if (i < 2 * NW + HQ * XK) {
        int j = i - 2 * NW;
        int n = j / XK, k = j % XK;
        float v = (k < OUTQ + CTXQ) ? Win[n * (OUTQ + CTXQ) + k] : 0.0f;
        Win_b[j] = (__bf16)v;
    }
}

// ---------------- h0/c0 init: context @ W^T + b, with reference's flat reshape ----------------
// h0(l,b,h) = Mflat[l*B*H + b*H + h], Mflat[i] = M[i/512][i%512], M = ctx @ W^T + b
__global__ void init_state(const float* __restrict__ z, const float* __restrict__ cond,
                           const float* __restrict__ Wh, const float* __restrict__ bh,
                           const float* __restrict__ Wc, const float* __restrict__ bc,
                           float* __restrict__ h0, float* __restrict__ c0) {
    int i = blockIdx.x * blockDim.x + threadIdx.x;
    const int N = LQ * BQ * HQ;  // 262144
    if (i >= 2 * N) return;
    bool isc = (i >= N);
    int j = isc ? i - N : i;
    int row = j >> 9;        // / (L*H)=512
    int col = j & 511;
    const float* W  = isc ? Wc : Wh;
    const float* bb = isc ? bc : bh;
    float acc = bb[col];
    const float* zr = z + row * LATQ;
    const float* cr = cond + row * CONDQ;
    const float* wr = W + col * CTXQ;
    for (int k = 0; k < LATQ;  ++k) acc += zr[k] * wr[k];
    for (int k = 0; k < CONDQ; ++k) acc += cr[k] * wr[LATQ + k];
    (isc ? c0 : h0)[j] = acc;
}

// ---------------- persistent LSTM kernel: 32 WGs x 16 batch rows, 16 waves ----------------
// Wave w owns a 16-wide h-slice (cols w*16..w*16+15): one WMMA N-tile per gate
// (i,f,g,o) -> acc[4] (32 VGPRs), cell state creg[2][8] (16 VGPRs). Cell update is
// register-only; only h goes to LDS (bf16) for the K-dimension of the next matmul.
__global__ __launch_bounds__(NT) void lstm_persistent(
    const __bf16* __restrict__ Wih_b, const __bf16* __restrict__ Whh_b,
    const __bf16* __restrict__ Win_b,
    const float* __restrict__ b_in, const float* __restrict__ b_ih, const float* __restrict__ b_hh,
    const float* __restrict__ W_out, const float* __restrict__ b_out,
    const float* __restrict__ z, const float* __restrict__ cond,
    const float* __restrict__ h0, const float* __restrict__ c0,
    const int* __restrict__ steps_p, float* __restrict__ out)
{
    __shared__ __align__(16) __bf16 inbuf[BT * INST];        // [prev(2) | z(64) | cond(32) | 0-pad]
    __shared__ __align__(16) __bf16 xbuf[BT * HST];          // current layer input (bf16)
    __shared__ __align__(16) __bf16 hbuf[LQ][BT * HST];      // hidden state per layer (bf16)
    __shared__ float ybuf[BT][OUTQ][8];                      // output-head partial sums

    const int tid   = threadIdx.x;
    const int wave  = tid >> 5;          // 0..15
    const int lane  = tid & 31;
    const int row16 = lane & 15;
    const int half  = lane >> 4;
    const int b0    = blockIdx.x * BT;
    const int T     = steps_p[0];
    const int colc  = wave * 16 + row16; // this lane's output column within H

    // time-invariant per-lane fragment offsets
    const int aofsI = row16 * INST + half * 16;  // A-frag base into inbuf
    const int aofsH = row16 * HST  + half * 16;  // A-frag base into xbuf/hbuf
    const int xbofs = colc * XK + half * 16;     // B-frag base into Win_b
    int bofs[4];                                  // B-frag bases into Wih/Whh (per gate)
    #pragma unroll
    for (int gi = 0; gi < 4; ++gi) bofs[gi] = (gi * HQ + colc) * HQ + half * 16;
    const int cofs = half * 8 * HST + colc;      // C-frag store base (row r -> + r*HST)

    // build inbuf (prev0 = 0)
    for (int i = tid; i < BT * INST; i += NT) {
        int r = i / INST, k = i % INST;
        float v = 0.f;
        if (k >= 2 && k < 2 + LATQ)             v = z[(b0 + r) * LATQ + (k - 2)];
        else if (k >= 2 + LATQ && k < 2 + CTXQ) v = cond[(b0 + r) * CONDQ + (k - 2 - LATQ)];
        inbuf[i] = (__bf16)v;
    }
    // load h0 -> hbuf (bf16)
    for (int i = tid; i < LQ * BT * HQ; i += NT) {
        int l = i / (BT * HQ);
        int rem = i % (BT * HQ);
        int r = rem / HQ, k = rem % HQ;
        hbuf[l][r * HST + k] = (__bf16)h0[l * (BQ * HQ) + (b0 + r) * HQ + k];
    }
    // cell state in registers, C-fragment layout: creg[l][r] ~ (row=r+half*8, col=colc)
    float creg[LQ][8];
    #pragma unroll
    for (int l = 0; l < LQ; ++l)
        #pragma unroll
        for (int r = 0; r < 8; ++r)
            creg[l][r] = c0[l * (BQ * HQ) + (b0 + r + half * 8) * HQ + colc];

    // hoisted biases (folded into WMMA accumulator init)
    float gbias[LQ][4], xbias;
    #pragma unroll
    for (int l = 0; l < LQ; ++l)
        #pragma unroll
        for (int gi = 0; gi < 4; ++gi)
            gbias[l][gi] = b_ih[l * GW + gi * HQ + colc] + b_hh[l * GW + gi * HQ + colc];
    xbias = b_in[colc];

    __syncthreads();

    #pragma unroll 1
    for (int ts = 0; ts < T; ++ts) {
        const int zofs = opaque_zero();   // blocks cross-step hoist of weight loads

        // ---- x = concat(prev, ctx) @ W_in^T + b_in  (WMMA, K=128 padded) ----
        const __bf16* WinP = Win_b + zofs;
        v8f xacc;
        #pragma unroll
        for (int r = 0; r < 8; ++r) xacc[r] = xbias;
        #pragma unroll 2
        for (int kc = 0; kc < 4; ++kc) {
            int k0 = kc * 32;
            v16bf a   = loadA(&inbuf[aofsI + k0]);
            v16bf bfr = *(const v16bf*)(WinP + xbofs + k0);
            xacc = wmma_bf16(a, bfr, xacc);
        }
        // previous step's xbuf readers finished at end-of-iter barrier
        #pragma unroll
        for (int r = 0; r < 8; ++r)
            xbuf[cofs + r * HST] = (__bf16)xacc[r];
        __syncthreads();

        // ---- two LSTM layers ----
        #pragma unroll
        for (int l = 0; l < LQ; ++l) {
            const __bf16* Wihl = Wih_b + l * (GW * HQ) + zofs;
            const __bf16* Whhl = Whh_b + l * (GW * HQ) + zofs;
            v8f acc[4];  // one N-tile per gate [i f g o] for this wave's 16-wide h-slice
            #pragma unroll
            for (int gi = 0; gi < 4; ++gi)
                #pragma unroll
                for (int r = 0; r < 8; ++r) acc[gi][r] = gbias[l][gi];
            // x @ W_ih^T  (K = 256)
            #pragma unroll 2
            for (int kc = 0; kc < 8; ++kc) {
                int k0 = kc * 32;
                v16bf a = loadA(&xbuf[aofsH + k0]);
                #pragma unroll
                for (int gi = 0; gi < 4; ++gi) {
                    v16bf bfr = *(const v16bf*)(Wihl + bofs[gi] + k0);
                    acc[gi] = wmma_bf16(a, bfr, acc[gi]);
                }
            }
            // h @ W_hh^T  (K = 256)
            #pragma unroll 2
            for (int kc = 0; kc < 8; ++kc) {
                int k0 = kc * 32;
                v16bf a = loadA(&hbuf[l][aofsH + k0]);
                #pragma unroll
                for (int gi = 0; gi < 4; ++gi) {
                    v16bf bfr = *(const v16bf*)(Whhl + bofs[gi] + k0);
                    acc[gi] = wmma_bf16(a, bfr, acc[gi]);
                }
            }
            __syncthreads();  // all waves done reading xbuf / hbuf[l]
            // cell update (all in registers), write new h as bf16
            #pragma unroll
            for (int r = 0; r < 8; ++r) {
                float iv = acc[0][r], fv = acc[1][r], gv = acc[2][r], ov = acc[3][r];
                float cn = sigm(fv) * creg[l][r] + sigm(iv) * tanh_(gv);
                float hn = sigm(ov) * tanh_(cn);
                creg[l][r] = cn;
                __bf16 hb = (__bf16)hn;
                int idx = cofs + r * HST;
                hbuf[l][idx] = hb;   // recurrent state for next step
                xbuf[idx]    = hb;   // input to next layer / output head
            }
            __syncthreads();
        }

        // ---- y = h_l1 @ W_out^T + b_out (16x2): parallel partials + LDS reduce ----
        if (tid < 256) {
            int r = tid & 15, o = (tid >> 4) & 1, ch = tid >> 5;  // 8 K-chunks of 32
            float p = 0.f;
            int kb = ch * 32;
            #pragma unroll 4
            for (int k = 0; k < 32; ++k)
                p += (float)xbuf[r * HST + kb + k] * W_out[o * HQ + kb + k];
            ybuf[r][o][ch] = p;
        }
        __syncthreads();
        if (tid < 32) {
            int r = tid & 15, o = tid >> 4;
            float accy = b_out[o];
            #pragma unroll
            for (int ch = 0; ch < 8; ++ch) accy += ybuf[r][o][ch];
            out[((long)(b0 + r) * T + ts) * OUTQ + o] = accy;
            inbuf[r * INST + o] = (__bf16)accy;   // feed back prev
        }
        __syncthreads();
    }
}

extern "C" void kernel_launch(void* const* d_in, const int* in_sizes, int n_in,
                              void* d_out, int out_size, void* d_ws, size_t ws_size,
                              hipStream_t stream) {
    const float* z    = (const float*)d_in[0];
    const float* cnd  = (const float*)d_in[1];
    const float* Wh   = (const float*)d_in[2];
    const float* bh   = (const float*)d_in[3];
    const float* Wc   = (const float*)d_in[4];
    const float* bc   = (const float*)d_in[5];
    const float* Win  = (const float*)d_in[6];
    const float* bin  = (const float*)d_in[7];
    const float* Wih  = (const float*)d_in[8];
    const float* Whh  = (const float*)d_in[9];
    const float* bih  = (const float*)d_in[10];
    const float* bhh  = (const float*)d_in[11];
    const float* Wout = (const float*)d_in[12];
    const float* bout = (const float*)d_in[13];
    const int*  steps = (const int*)d_in[14];

    // workspace layout (~4.26 MB)
    char* ws = (char*)d_ws;
    __bf16* Wih_b = (__bf16*)ws;                               // 524288 bf16
    __bf16* Whh_b = Wih_b + (size_t)LQ * GW * HQ;              // 524288 bf16
    __bf16* Win_b = Whh_b + (size_t)LQ * GW * HQ;              // 32768 bf16
    float*  h0w   = (float*)(Win_b + HQ * XK);                 // 262144 f32
    float*  c0w   = h0w + (size_t)LQ * BQ * HQ;                // 262144 f32

    const int convN = 2 * LQ * GW * HQ + HQ * XK;              // 1081344
    convert_weights<<<(convN + 255) / 256, 256, 0, stream>>>(Wih, Whh, Win, Wih_b, Whh_b, Win_b);
    init_state<<<(2 * LQ * BQ * HQ + 255) / 256, 256, 0, stream>>>(z, cnd, Wh, bh, Wc, bc, h0w, c0w);
    lstm_persistent<<<BQ / BT, NT, 0, stream>>>(Wih_b, Whh_b, Win_b, bin, bih, bhh,
                                                Wout, bout, z, cnd, h0w, c0w, steps,
                                                (float*)d_out);
}
